// HSVRNNBVH_29858612642194
// MI455X (gfx1250) — matrix operand order repaired and checked by reading, not compiled, not run on hardware
//
#include <hip/hip_runtime.h>
#include <hip/hip_bf16.h>
#include <cstdint>

// Problem constants (from reference)
#define B_      256
#define T_      128
#define K_      24
#define S_      10
#define L_      32
#define H_      512
#define STATE_  96            // K*(D+1)
#define SB_     (S_*B_)       // 2560
#define RI_N    64            // ri head output cols padded 27 -> 64
#define JM_N    192           // jm head output cols padded 144 -> 192

typedef _Float16 half16_t __attribute__((ext_vector_type(16)));
typedef _Float16 half8_t  __attribute__((ext_vector_type(8)));
typedef float    float8_t __attribute__((ext_vector_type(8)));

__device__ __forceinline__ float sigmoidf_(float x){ return 1.0f/(1.0f+__expf(-x)); }
__device__ __forceinline__ float softplusf_(float x){ return x>20.0f ? x : log1pf(__expf(x)); }

// ---------------------------------------------------------------------------
// Fragment load for one K-group of 32: A (2 x b128) + 4 B fragments (8 x b128).
// X/W stay wave-uniform bases (SGPR pair, saddr form); aoff/boff are the only
// lane-varying 32-bit offsets; i*ldw16 and k0 are uniform and fold into
// SGPR/immediate offsets -> minimal VGPR pressure, no 64-bit VALU addressing.
// ---------------------------------------------------------------------------
__device__ __forceinline__ void load_frag(const _Float16* __restrict__ X, int aoff,
                                          const _Float16* __restrict__ W, int boff,
                                          int ldw16, int k0,
                                          half16_t& a, half16_t b[4])
{
    ((half8_t*)&a)[0] = *(const half8_t*)(X + aoff + k0);
    ((half8_t*)&a)[1] = *(const half8_t*)(X + aoff + k0 + 8);
#pragma unroll
    for (int i = 0; i < 4; ++i) {
        ((half8_t*)&b[i])[0] = *(const half8_t*)(W + boff + i*ldw16 + k0);
        ((half8_t*)&b[i])[1] = *(const half8_t*)(W + boff + i*ldw16 + k0 + 8);
    }
}

__device__ __forceinline__ void wmma4(float8_t c[4], const half16_t& a, const half16_t b[4])
{
#pragma unroll
    for (int i = 0; i < 4; ++i)
        c[i] = __builtin_amdgcn_wmma_f32_16x16x32_f16(false, a, false, b[i],
                                                      (short)0, c[i], false, false);
}

// ---------------------------------------------------------------------------
// f16 WMMA GEMM:  out = act( X(MxK) * W(NxK)^T + bias ),  N % 64 == 0, K % 32 == 0
// One wave computes a 16x64 output tile (4 independent accumulator chains).
// K loop is explicitly double-buffered: next K-group's loads are issued as a
// clause before the current group's 4 WMMAs, so waits are partial and load
// latency overlaps matrix work.
// act: 0=none, 1=leaky(0.01), 2=tanh
// ---------------------------------------------------------------------------
__global__ void gemm_wmma(const _Float16* __restrict__ X, int ldx,
                          const _Float16* __restrict__ W, int ldw,
                          const float* __restrict__ bias,
                          float* __restrict__ outF, _Float16* __restrict__ outH,
                          int M, int N, int K, int act)
{
    const int lane = threadIdx.x & 31;
    const int wave = threadIdx.x >> 5;
    const int ngrp = N >> 6;                       // groups of 4 N-tiles
    const int g = blockIdx.x * (blockDim.x >> 5) + wave;
    if (g >= (M >> 4) * ngrp) return;
    const int tm  = g / ngrp;
    const int tn0 = (g % ngrp) * 4;

    const int hsel = lane >> 4;                    // 0: lanes 0-15, 1: lanes 16-31
    const int l16  = lane & 15;

    // A: lane holds row (tm*16+l16); 16 contiguous halves at K k0+16*hsel
    const int aoff = (tm*16 + l16) * ldx + (hsel ? 16 : 0);
    // B[i]: lane holds W row ((tn0+i)*16+l16); 16 contiguous halves at K k0+16*hsel
    const int boff = (tn0*16 + l16) * ldw + (hsel ? 16 : 0);
    const int ldw16 = 16 * ldw;                    // uniform stride between B tiles

    float8_t c[4];
#pragma unroll
    for (int i = 0; i < 4; ++i) c[i] = (float8_t){0.f,0.f,0.f,0.f,0.f,0.f,0.f,0.f};

    const int G = K >> 5;                          // number of K-groups of 32
    half16_t aA, bA[4], aB, bB[4];
    load_frag(X, aoff, W, boff, ldw16, 0, aA, bA); // prologue: group 0 -> A
    int gk = 0;
    for (; gk + 2 <= G; gk += 2) {
        load_frag(X, aoff, W, boff, ldw16, (gk + 1) << 5, aB, bB); // prefetch gk+1
        wmma4(c, aA, bA);                                          // consume gk
        if (gk + 2 < G)
            load_frag(X, aoff, W, boff, ldw16, (gk + 2) << 5, aA, bA); // prefetch gk+2
        wmma4(c, aB, bB);                                          // consume gk+1
    }
    if (gk < G) wmma4(c, aA, bA);                  // odd-G epilogue (already loaded)

#pragma unroll
    for (int i = 0; i < 4; ++i) {
        const int col = (tn0 + i)*16 + l16;
        const float bv = bias ? bias[col] : 0.0f;
#pragma unroll
        for (int j = 0; j < 8; ++j) {
            float v = c[i][j] + bv;
            if (act == 1) v = (v > 0.0f) ? v : 0.01f * v;
            else if (act == 2) v = tanhf(v);
            const int row = tm*16 + hsel*8 + j;
            if (outF) outF[(size_t)row * N + col] = v;
            if (outH) outH[(size_t)row * N + col] = (_Float16)v;
        }
    }
}

// ---------------------------------------------------------------------------
// Weight f32 -> f16 conversion with optional zero row padding
// ---------------------------------------------------------------------------
__global__ void k_cvt_f16(const float* __restrict__ src, _Float16* __restrict__ dst,
                          int srcRows, int dstRows, int cols)
{
    int i = blockIdx.x * blockDim.x + threadIdx.x;
    int n = dstRows * cols;
    if (i >= n) return;
    int r = i / cols;
    dst[i] = (r < srcRows) ? (_Float16)src[i] : (_Float16)0.0f;
}

__global__ void k_pad_bias(const float* __restrict__ src, float* __restrict__ dst,
                           int n, int npad)
{
    int i = blockIdx.x * blockDim.x + threadIdx.x;
    if (i < npad) dst[i] = (i < n) ? src[i] : 0.0f;
}

// ---------------------------------------------------------------------------
// Precompute: per-(b,k) median parent distance over T and bone offsets
// ---------------------------------------------------------------------------
__global__ void k_offsets(const float* __restrict__ kp, const float* __restrict__ offp,
                          float* __restrict__ offset)
{
    int b = blockIdx.x;
    int k = threadIdx.x;
    if (k >= K_) return;
    int p = (k == 0) ? 0 : ((k - 1) >> 1);
    float dloc[T_];
    for (int t = 0; t < T_; ++t) {
        const float* a = kp + ((size_t)(b*T_ + t)*K_ + k) * 4;
        const float* c = kp + ((size_t)(b*T_ + t)*K_ + p) * 4;
        float dx = a[0]-c[0], dy = a[1]-c[1], dz = a[2]-c[2];
        dloc[t] = sqrtf(dx*dx + dy*dy + dz*dz);
    }
    // rank-63 selection (== jnp.sort(...)[63])
    float med = 0.0f;
    for (int i = 0; i < T_; ++i) {
        int lt = 0, eq = 0;
        float di = dloc[i];
        for (int j = 0; j < T_; ++j) { lt += (dloc[j] < di); eq += (dloc[j] == di); }
        if (lt <= 63 && 63 < lt + eq) { med = di; break; }
    }
    float ox = offp[k*3+0], oy = offp[k*3+1], oz = offp[k*3+2];
    float nn = sqrtf(ox*ox + oy*oy + oz*oz) + 1e-10f;
    offset[(b*K_ + k)*3 + 0] = ox / nn * med;
    offset[(b*K_ + k)*3 + 1] = oy / nn * med;
    offset[(b*K_ + k)*3 + 2] = oz / nn * med;
}

// ---------------------------------------------------------------------------
// Init: h0 broadcast, h_out[:,0,:], zero accumulators
// ---------------------------------------------------------------------------
__global__ void k_init(const float* __restrict__ init_state,
                       float* __restrict__ h_cur, float* __restrict__ h_out,
                       float* __restrict__ accs)
{
    int i = blockIdx.x * blockDim.x + threadIdx.x;
    if (i < 2) accs[i] = 0.0f;
    if (i >= B_*H_) return;
    int b = i / H_, hh = i % H_;
    float v = init_state[hh];
    h_cur[i] = v;
    h_out[(size_t)b * (T_+1) * H_ + hh] = v;
}

// ---------------------------------------------------------------------------
// Per-step prep: h16 = f16(h_cur); post_in = f16(concat(h, kp_t))
// ---------------------------------------------------------------------------
__global__ void k_prep(const float* __restrict__ h_cur, const float* __restrict__ keypoints,
                       int t, _Float16* __restrict__ h16, _Float16* __restrict__ post_in)
{
    int i = blockIdx.x * blockDim.x + threadIdx.x;
    if (i >= B_ * (H_ + STATE_)) return;
    int b = i / (H_ + STATE_), col = i % (H_ + STATE_);
    if (col < H_) {
        _Float16 v = (_Float16)h_cur[(size_t)b*H_ + col];
        h16[(size_t)b*H_ + col] = v;
        post_in[i] = v;
    } else {
        post_in[i] = (_Float16)keypoints[(size_t)(b*T_ + t)*STATE_ + (col - H_)];
    }
}

// ---------------------------------------------------------------------------
// Post-process prior/post heads: softplus, KL accumulation (block reduce+atomic)
// ---------------------------------------------------------------------------
__global__ void k_postproc(const float* __restrict__ pr, const float* __restrict__ po,
                           float* __restrict__ qm, float* __restrict__ qs,
                           float* __restrict__ klacc)
{
    int i = blockIdx.x * blockDim.x + threadIdx.x;
    float kl = 0.0f;
    if (i < B_ * L_) {
        int b = i / L_, l = i % L_;
        float pm  = pr[b*64 + l];
        float ps  = softplusf_(pr[b*64 + 32 + l]) + 1e-4f;
        float qmv = po[b*64 + l];
        float qsv = softplusf_(po[b*64 + 32 + l]) + 1e-4f;
        qm[i] = qmv; qs[i] = qsv;
        float dm = qmv - pm;
        kl = __logf(ps / qsv) + (qsv*qsv + dm*dm) / (2.0f*ps*ps) - 0.5f;
    }
    __shared__ float red[256];
    red[threadIdx.x] = kl; __syncthreads();
    for (int s = 128; s > 0; s >>= 1) {
        if (threadIdx.x < s) red[threadIdx.x] += red[threadIdx.x + s];
        __syncthreads();
    }
    if (threadIdx.x == 0) atomicAdd(klacc, red[0]);
}

// ---------------------------------------------------------------------------
// Build decoder input (S*B x 544 f16) and z (f32)
// ---------------------------------------------------------------------------
__global__ void k_build_decin(const _Float16* __restrict__ h16,
                              const float* __restrict__ qm, const float* __restrict__ qs,
                              const float* __restrict__ eps, int t,
                              _Float16* __restrict__ decin, float* __restrict__ z)
{
    int i = blockIdx.x * blockDim.x + threadIdx.x;
    const int W = H_ + L_;             // 544
    if (i >= SB_ * W) return;
    int row = i / W, col = i % W;
    int s = row / B_, b = row % B_;
    if (col < H_) {
        decin[i] = h16[(size_t)b*H_ + col];
    } else {
        int l = col - H_;
        float zv = qm[b*L_ + l] + qs[b*L_ + l] * eps[((size_t)(t*S_ + s)*B_ + b)*L_ + l];
        z[(size_t)row*L_ + l] = zv;
        decin[i] = (_Float16)zv;
    }
}

// ---------------------------------------------------------------------------
// rot6d -> R, FK chain, kp candidates, distance to observation
// one thread per (s,b) row; Rg/p staged in LDS
// raw has row stride RI_N, jm has row stride JM_N (padded GEMM outputs)
// ---------------------------------------------------------------------------
__global__ void k_fk(const float* __restrict__ raw,      // SB x RI_N (tanh'd)
                     const float* __restrict__ jm,       // SB x JM_N
                     const float* __restrict__ offset,   // B x K x 3
                     const float* __restrict__ keypoints, int t,
                     float* __restrict__ kp_s,           // SB x 96
                     float* __restrict__ R_s,            // SB x 216
                     float* __restrict__ dist)           // SB
{
    __shared__ float sR[32][K_][9];
    __shared__ float sP[32][K_][3];
    int tid = threadIdx.x;
    int row = blockIdx.x * 32 + tid;           // row = s*B + b
    if (row >= SB_) return;
    int b = row % B_;
    const float* rw = raw + (size_t)row * RI_N;
    const float* jr = jm  + (size_t)row * JM_N;

    for (int k = 0; k < K_; ++k) {
        float x0 = jr[k*6+0], x1 = jr[k*6+1], x2 = jr[k*6+2];
        float x3 = jr[k*6+3], x4 = jr[k*6+4], x5 = jr[k*6+5];
        float n1 = sqrtf(x0*x0 + x1*x1 + x2*x2) + 1e-8f;
        float b1x = x0/n1, b1y = x1/n1, b1z = x2/n1;
        float dp = b1x*x3 + b1y*x4 + b1z*x5;
        float a2x = x3 - dp*b1x, a2y = x4 - dp*b1y, a2z = x5 - dp*b1z;
        float n2 = sqrtf(a2x*a2x + a2y*a2y + a2z*a2z) + 1e-8f;
        float b2x = a2x/n2, b2y = a2y/n2, b2z = a2z/n2;
        float b3x = b1y*b2z - b1z*b2y;
        float b3y = b1z*b2x - b1x*b2z;
        float b3z = b1x*b2y - b1y*b2x;
        float Rl[9] = {b1x,b1y,b1z, b2x,b2y,b2z, b3x,b3y,b3z};
        if (k == 0) {
            for (int j = 0; j < 9; ++j) sR[tid][0][j] = Rl[j];
            sP[tid][0][0] = rw[0]; sP[tid][0][1] = rw[1]; sP[tid][0][2] = rw[2];
        } else {
            int p = (k - 1) >> 1;
            float Rg[9];
            for (int r = 0; r < 3; ++r)
                for (int c = 0; c < 3; ++c)
                    Rg[r*3+c] = sR[tid][p][r*3+0]*Rl[0*3+c]
                              + sR[tid][p][r*3+1]*Rl[1*3+c]
                              + sR[tid][p][r*3+2]*Rl[2*3+c];
            for (int j = 0; j < 9; ++j) sR[tid][k][j] = Rg[j];
            const float* of = offset + (size_t)(b*K_ + k)*3;
            for (int r = 0; r < 3; ++r)
                sP[tid][k][r] = Rg[r*3+0]*of[0] + Rg[r*3+1]*of[1] + Rg[r*3+2]*of[2]
                              + sP[tid][p][r];
        }
    }

    const float* kpf = keypoints + (size_t)(b*T_ + t) * STATE_;
    float dsum = 0.0f;
    for (int k = 0; k < K_; ++k) {
        float inten = (rw[3 + k] + 1.0f) * 0.5f;
        float px = sP[tid][k][0], py = sP[tid][k][1], pz = sP[tid][k][2];
        kp_s[(size_t)row*STATE_ + k*4 + 0] = px;
        kp_s[(size_t)row*STATE_ + k*4 + 1] = py;
        kp_s[(size_t)row*STATE_ + k*4 + 2] = pz;
        kp_s[(size_t)row*STATE_ + k*4 + 3] = inten;
        for (int j = 0; j < 9; ++j) R_s[(size_t)row*216 + k*9 + j] = sR[tid][k][j];
        float d0 = kpf[k*4+0] - px, d1 = kpf[k*4+1] - py;
        float d2 = kpf[k*4+2] - pz, d3 = kpf[k*4+3] - inten;
        dsum += d0*d0 + d1*d1 + d2*d2 + d3*d3;
    }
    dist[row] = dsum;
}

// ---------------------------------------------------------------------------
// argmin over S, gather best candidate, write step outputs + GRU input (f16)
// ---------------------------------------------------------------------------
__global__ void k_select(const float* __restrict__ dist, const float* __restrict__ kp_s,
                         const float* __restrict__ R_s, const float* __restrict__ z,
                         float* __restrict__ kp_out, float* __restrict__ R_out,
                         float* __restrict__ z_out, _Float16* __restrict__ grux, int t)
{
    int b = blockIdx.x;
    __shared__ int sIdx;
    if (threadIdx.x == 0) {
        float best = dist[b]; int bi = 0;
        for (int s = 1; s < S_; ++s) {
            float v = dist[s*B_ + b];
            if (v < best) { best = v; bi = s; }
        }
        sIdx = bi;
    }
    __syncthreads();
    int row = sIdx * B_ + b;
    for (int j = threadIdx.x; j < STATE_; j += 64) {
        float v = kp_s[(size_t)row*STATE_ + j];
        kp_out[(size_t)(b*T_ + t)*STATE_ + j] = v;
        grux[b*128 + j] = (_Float16)v;
    }
    for (int j = threadIdx.x; j < 216; j += 64)
        R_out[(size_t)(b*T_ + t)*216 + j] = R_s[(size_t)row*216 + j];
    for (int j = threadIdx.x; j < L_; j += 64) {
        float v = z[(size_t)row*L_ + j];
        z_out[(size_t)(b*T_ + t)*L_ + j] = v;
        grux[b*128 + STATE_ + j] = (_Float16)v;
    }
}

// ---------------------------------------------------------------------------
// GRU gate fusion + state update
// ---------------------------------------------------------------------------
__global__ void k_gru(const float* __restrict__ gi, const float* __restrict__ gh,
                      float* __restrict__ h_cur, float* __restrict__ h_out, int t)
{
    int i = blockIdx.x * blockDim.x + threadIdx.x;
    if (i >= B_*H_) return;
    int b = i / H_, hh = i % H_;
    const float* gib = gi + (size_t)b * 3*H_;
    const float* ghb = gh + (size_t)b * 3*H_;
    float r = sigmoidf_(gib[hh]        + ghb[hh]);
    float u = sigmoidf_(gib[H_ + hh]   + ghb[H_ + hh]);
    float n = tanhf(gib[2*H_ + hh] + r * ghb[2*H_ + hh]);
    float hp = h_cur[i];
    float hv = (1.0f - u) * n + u * hp;
    h_cur[i] = hv;
    h_out[(size_t)b*(T_+1)*H_ + (size_t)(t+1)*H_ + hh] = hv;
}

// ---------------------------------------------------------------------------
// recon = sum((kp_out - keypoints)^2) accumulation; then finalize scalars
// ---------------------------------------------------------------------------
__global__ void k_recon(const float* __restrict__ kp_out, const float* __restrict__ keypoints,
                        float* __restrict__ racc)
{
    int i = blockIdx.x * blockDim.x + threadIdx.x;
    float v = 0.0f;
    if (i < B_*T_*K_*4) {
        float d = kp_out[i] - keypoints[i];
        v = d * d;
    }
    __shared__ float red[256];
    red[threadIdx.x] = v; __syncthreads();
    for (int s = 128; s > 0; s >>= 1) {
        if (threadIdx.x < s) red[threadIdx.x] += red[threadIdx.x + s];
        __syncthreads();
    }
    if (threadIdx.x == 0) atomicAdd(racc, red[0]);
}

__global__ void k_final(const float* __restrict__ accs, float* __restrict__ out_kl,
                        float* __restrict__ out_rec)
{
    if (threadIdx.x == 0 && blockIdx.x == 0) {
        *out_kl  = accs[0] / (float)(T_ * B_ * L_);
        *out_rec = accs[1] / (float)(B_ * T_);
    }
}

// ===========================================================================
// Host launcher
// ===========================================================================
static inline char* ws_take(char*& cur, size_t bytes) {
    char* p = cur;
    cur += (bytes + 255) & ~(size_t)255;
    return p;
}

extern "C" void kernel_launch(void* const* d_in, const int* in_sizes, int n_in,
                              void* d_out, int out_size, void* d_ws, size_t ws_size,
                              hipStream_t stream)
{
    (void)in_sizes; (void)n_in; (void)out_size; (void)ws_size;
    const float* keypoints  = (const float*)d_in[0];
    const float* eps        = (const float*)d_in[1];
    const float* post_w1    = (const float*)d_in[2];
    const float* post_b1    = (const float*)d_in[3];
    const float* post_w2    = (const float*)d_in[4];
    const float* post_b2    = (const float*)d_in[5];
    const float* prior_w1   = (const float*)d_in[6];
    const float* prior_b1   = (const float*)d_in[7];
    const float* prior_w2   = (const float*)d_in[8];
    const float* prior_b2   = (const float*)d_in[9];
    const float* ri_w1      = (const float*)d_in[10];
    const float* ri_b1      = (const float*)d_in[11];
    const float* ri_w2      = (const float*)d_in[12];
    const float* ri_b2      = (const float*)d_in[13];
    const float* jm_w1      = (const float*)d_in[14];
    const float* jm_b1      = (const float*)d_in[15];
    const float* jm_w2      = (const float*)d_in[16];
    const float* jm_b2      = (const float*)d_in[17];
    const float* gru_w_ih   = (const float*)d_in[18];
    const float* gru_b_ih   = (const float*)d_in[19];
    const float* gru_w_hh   = (const float*)d_in[20];
    const float* gru_b_hh   = (const float*)d_in[21];
    const float* init_state = (const float*)d_in[22];
    const float* offset_par = (const float*)d_in[23];
    // d_in[24] = parents (always (k-1)/2, hardcoded)

    // ---- output layout (f32, concatenated) ----
    float* out     = (float*)d_out;
    float* kp_out  = out;                                     // B*T*K*4
    float* R_out   = kp_out + (size_t)B_*T_*K_*4;             // B*T*K*9
    float* z_out   = R_out  + (size_t)B_*T_*K_*9;             // B*T*L
    float* h_out   = z_out  + (size_t)B_*T_*L_;               // B*(T+1)*H
    float* kl_out  = h_out  + (size_t)B_*(T_+1)*H_;           // scalar
    float* rec_out = kl_out + 1;                              // scalar

    // ---- workspace layout (~18 MB) ----
    char* cur = (char*)d_ws;
    _Float16* w_post1  = (_Float16*)ws_take(cur, sizeof(_Float16)*128*608);
    _Float16* w_post2  = (_Float16*)ws_take(cur, sizeof(_Float16)*64*128);
    _Float16* w_prior1 = (_Float16*)ws_take(cur, sizeof(_Float16)*128*512);
    _Float16* w_prior2 = (_Float16*)ws_take(cur, sizeof(_Float16)*64*128);
    _Float16* w_ri1    = (_Float16*)ws_take(cur, sizeof(_Float16)*128*544);
    _Float16* w_ri2    = (_Float16*)ws_take(cur, sizeof(_Float16)*RI_N*128);  // 27 -> 64
    _Float16* w_jm1    = (_Float16*)ws_take(cur, sizeof(_Float16)*128*544);
    _Float16* w_jm2    = (_Float16*)ws_take(cur, sizeof(_Float16)*JM_N*128);  // 144 -> 192
    _Float16* w_gih    = (_Float16*)ws_take(cur, sizeof(_Float16)*1536*128);
    _Float16* w_ghh    = (_Float16*)ws_take(cur, sizeof(_Float16)*1536*512);
    float*    b_ri2p   = (float*)   ws_take(cur, sizeof(float)*RI_N);
    float*    b_jm2p   = (float*)   ws_take(cur, sizeof(float)*JM_N);
    float*    offset   = (float*)   ws_take(cur, sizeof(float)*B_*K_*3);
    float*    h_cur    = (float*)   ws_take(cur, sizeof(float)*B_*H_);
    float*    pr       = (float*)   ws_take(cur, sizeof(float)*B_*64);
    float*    po       = (float*)   ws_take(cur, sizeof(float)*B_*64);
    float*    qm       = (float*)   ws_take(cur, sizeof(float)*B_*L_);
    float*    qs       = (float*)   ws_take(cur, sizeof(float)*B_*L_);
    float*    zbuf     = (float*)   ws_take(cur, sizeof(float)*SB_*L_);
    float*    raw      = (float*)   ws_take(cur, sizeof(float)*SB_*RI_N);
    float*    jmout    = (float*)   ws_take(cur, sizeof(float)*SB_*JM_N);
    float*    kp_s     = (float*)   ws_take(cur, sizeof(float)*SB_*STATE_);
    float*    R_s      = (float*)   ws_take(cur, sizeof(float)*SB_*216);
    float*    dist     = (float*)   ws_take(cur, sizeof(float)*SB_);
    float*    gi       = (float*)   ws_take(cur, sizeof(float)*B_*3*H_);
    float*    gh       = (float*)   ws_take(cur, sizeof(float)*B_*3*H_);
    float*    accs     = (float*)   ws_take(cur, sizeof(float)*2);            // kl, recon
    _Float16* h16      = (_Float16*)ws_take(cur, sizeof(_Float16)*B_*H_);
    _Float16* post_in  = (_Float16*)ws_take(cur, sizeof(_Float16)*B_*(H_+STATE_));
    _Float16* posthid  = (_Float16*)ws_take(cur, sizeof(_Float16)*B_*128);
    _Float16* priorhid = (_Float16*)ws_take(cur, sizeof(_Float16)*B_*128);
    _Float16* decin    = (_Float16*)ws_take(cur, sizeof(_Float16)*SB_*(H_+L_));
    _Float16* rihid    = (_Float16*)ws_take(cur, sizeof(_Float16)*SB_*128);
    _Float16* jmhid    = (_Float16*)ws_take(cur, sizeof(_Float16)*SB_*128);
    _Float16* grux     = (_Float16*)ws_take(cur, sizeof(_Float16)*B_*128);

    auto cvt = [&](const float* s, _Float16* d, int rows, int rowsPad, int cols) {
        int n = rowsPad * cols;
        k_cvt_f16<<<(n + 255)/256, 256, 0, stream>>>(s, d, rows, rowsPad, cols);
    };
    auto gemm = [&](const _Float16* X, int ldx, const _Float16* Wp, int ldw,
                    const float* bias, float* oF, _Float16* oH,
                    int M, int N, int Kd, int act) {
        int groups = (M/16) * (N/64);        // one wave per 16x64 tile group
        gemm_wmma<<<(groups + 3)/4, 128, 0, stream>>>(X, ldx, Wp, ldw, bias, oF, oH,
                                                      M, N, Kd, act);
    };

    // ---- one-time (per call) setup ----
    cvt(post_w1,  w_post1,  128, 128,  608);
    cvt(post_w2,  w_post2,   64,  64,  128);
    cvt(prior_w1, w_prior1, 128, 128,  512);
    cvt(prior_w2, w_prior2,  64,  64,  128);
    cvt(ri_w1,    w_ri1,    128, 128,  544);
    cvt(ri_w2,    w_ri2,     27, RI_N, 128);
    cvt(jm_w1,    w_jm1,    128, 128,  544);
    cvt(jm_w2,    w_jm2,    144, JM_N, 128);
    cvt(gru_w_ih, w_gih,   1536,1536,  128);
    cvt(gru_w_hh, w_ghh,   1536,1536,  512);
    k_pad_bias<<<1, RI_N, 0, stream>>>(ri_b2, b_ri2p, 27, RI_N);
    k_pad_bias<<<1, JM_N, 0, stream>>>(jm_b2, b_jm2p, 144, JM_N);
    k_offsets<<<B_, 32, 0, stream>>>(keypoints, offset_par, offset);
    k_init<<<(B_*H_ + 255)/256, 256, 0, stream>>>(init_state, h_cur, h_out, accs);

    // ---- sequential scan over T ----
    for (int t = 0; t < T_; ++t) {
        k_prep<<<(B_*(H_+STATE_) + 255)/256, 256, 0, stream>>>(h_cur, keypoints, t,
                                                               h16, post_in);
        // posterior / prior MLPs
        gemm(post_in, 608, w_post1, 608, post_b1, nullptr, posthid, B_, 128, 608, 1);
        gemm(posthid, 128, w_post2, 128, post_b2, po, nullptr,      B_,  64, 128, 0);
        gemm(h16,     512, w_prior1,512, prior_b1, nullptr, priorhid,B_, 128, 512, 1);
        gemm(priorhid,128, w_prior2,128, prior_b2, pr, nullptr,     B_,  64, 128, 0);
        k_postproc<<<(B_*L_ + 255)/256, 256, 0, stream>>>(pr, po, qm, qs, accs);
        // z + decoder input
        k_build_decin<<<(SB_*(H_+L_) + 255)/256, 256, 0, stream>>>(h16, qm, qs, eps, t,
                                                                   decin, zbuf);
        // decoder MLPs (S*B rows)
        gemm(decin, 544, w_ri1, 544, ri_b1, nullptr, rihid, SB_, 128, 544, 1);
        gemm(rihid, 128, w_ri2, 128, b_ri2p, raw, nullptr,  SB_, RI_N, 128, 2);
        gemm(decin, 544, w_jm1, 544, jm_b1, nullptr, jmhid, SB_, 128, 544, 1);
        gemm(jmhid, 128, w_jm2, 128, b_jm2p, jmout, nullptr,SB_, JM_N, 128, 0);
        // FK + distances, argmin select
        k_fk<<<SB_/32, 32, 0, stream>>>(raw, jmout, offset, keypoints, t, kp_s, R_s, dist);
        k_select<<<B_, 64, 0, stream>>>(dist, kp_s, R_s, zbuf, kp_out, R_out, z_out,
                                        grux, t);
        // GRU
        gemm(grux, 128, w_gih, 128, gru_b_ih, gi, nullptr, B_, 3*H_, 128, 0);
        gemm(h16,  512, w_ghh, 512, gru_b_hh, gh, nullptr, B_, 3*H_, 512, 0);
        k_gru<<<(B_*H_ + 255)/256, 256, 0, stream>>>(gi, gh, h_cur, h_out, t);
    }

    // ---- final scalar reductions ----
    k_recon<<<(B_*T_*K_*4 + 255)/256, 256, 0, stream>>>(kp_out, keypoints, accs + 1);
    k_final<<<1, 64, 0, stream>>>(accs, kl_out, rec_out);
}
